// Repr1Classifier_25967372272284
// MI455X (gfx1250) — matrix operand order repaired and checked by reading, not compiled, not run on hardware
//
#include <hip/hip_runtime.h>
#include <hip/hip_bf16.h>

// ---------------------------------------------------------------------------
// Problem constants (match reference)
// ---------------------------------------------------------------------------
#define N_HOST 20000
#define N_FLOW 200000
#define E_HF   400000
#define E_REL  800000
#define GB     64      // graphs in batch
#define HD     128     // hidden dim
#define FI     97      // flow input dim
#define NC     10      // classes

typedef __attribute__((ext_vector_type(16))) __bf16 v16bf;
typedef __attribute__((ext_vector_type(8)))  float  v8f;

__device__ __forceinline__ __bf16 f2bf(float x) { return (__bf16)x; }

// ---------------------------------------------------------------------------
// Zero fill / elementwise add
// ---------------------------------------------------------------------------
__global__ void fill_zero_f(float* __restrict__ p, long long n) {
    long long i = (long long)blockIdx.x * blockDim.x + threadIdx.x;
    if (i < n) p[i] = 0.0f;
}
__global__ void fill_zero_u(unsigned* __restrict__ p, long long n) {
    long long i = (long long)blockIdx.x * blockDim.x + threadIdx.x;
    if (i < n) p[i] = 0u;
}
__global__ void add2_f(const float* __restrict__ a, const float* __restrict__ b,
                       float* __restrict__ o, int n) {
    int i = blockIdx.x * blockDim.x + threadIdx.x;
    if (i < n) o[i] = a[i] + b[i];
}

// pad [N x FI] (stride FI) -> [N x HD] (stride HD), zero cols FI..HD-1
__global__ void pad_flow(const float* __restrict__ in, float* __restrict__ out) {
    int n = blockIdx.x;
    int c = threadIdx.x;     // 128
    out[(long long)n * HD + c] = (c < FI) ? in[(long long)n * FI + c] : 0.0f;
}

// ---------------------------------------------------------------------------
// Embedding gather: out[n, 0:128] = table[ids[n], 0:128]
// ---------------------------------------------------------------------------
__global__ void gather_embed(const float* __restrict__ table,
                             const int* __restrict__ ids,
                             float* __restrict__ out) {
    int n = blockIdx.x;
    int c = threadIdx.x;     // 128
    out[(long long)n * HD + c] = table[(long long)ids[n] * HD + c];
}

// ---------------------------------------------------------------------------
// Edge scatter-add (segment_sum): agg[dst[e], c] += feat[src[e], c]
// feat has row stride fstride; agg has row stride HD. One block per edge:
// src/dst become scalar loads, feature row access fully coalesced.
// ---------------------------------------------------------------------------
__global__ void scatter_add(const float* __restrict__ feat, int fstride,
                            const int* __restrict__ src,
                            const int* __restrict__ dst,
                            int dim, float* __restrict__ agg) {
    int e = blockIdx.x;
    int c = threadIdx.x;     // 128
    if (c >= dim) return;
    int s = src[e];
    int d = dst[e];
    atomicAdd(&agg[(long long)d * HD + c], feat[(long long)s * fstride + c]);
}

// ---------------------------------------------------------------------------
// Fused GEMM: out[N x 128] = relu?( sum_t A_t @ W_t + bias )
// A_t: [N x 128] f32 row-major (K<128 terms are zero-padded in memory).
// W_t: [K_t x 128] f32; rows >= K_t are zero-filled at LDS staging.
// 256 threads = 8 waves; block does 16 rows x 128 cols; wave w owns cols
// [16w,16w+16). K loop fully unrolled: 4 x v_wmma_f32_16x16x32_bf16.
// W staged in LDS pre-swizzled to B-fragment order: lane's 16 bf16 values
// are contiguous (one 32B read per fragment).
//   Ws[((kt4*8 + w)*32 + l)*16 + i] = bf16(W[kt4*32 + 8*(l>>4) + 16*(i>>3) + (i&7)][16w + (l&15)])
// ---------------------------------------------------------------------------
__device__ __forceinline__ void stage_w(const float* __restrict__ W, int K,
                                        __bf16* __restrict__ Ws) {
    const __bf16 zb = f2bf(0.0f);
    for (int src = threadIdx.x; src < HD * HD; src += 256) {
        int k = src >> 7;          // row of W
        int n = src & 127;         // col of W
        int l = (((k >> 3) & 1) << 4) | (n & 15);
        int d = ((((k >> 5) << 3) + (n >> 4)) * 32 + l) * 16 + ((k & 7) | (((k >> 4) & 1) << 3));
        Ws[d] = (k < K) ? f2bf(W[src]) : zb;
    }
}

__device__ __forceinline__ void gemm_term(const float* __restrict__ A,
                                          const __bf16* __restrict__ Ws,
                                          v8f& c, int lane, int wave, int row) {
    const float4* A4 = (const float4*)(A + (long long)row * HD);
    const v16bf*  Bp = (const v16bf*)Ws + (wave * 32 + lane);
    const int fo = (lane >> 4) << 1;          // float4 offset: 0 or 2 (=8 floats)
#pragma unroll
    for (int kt4 = 0; kt4 < 4; ++kt4) {
        int f = kt4 * 8 + fo;
        float4 x0 = A4[f];
        float4 x1 = A4[f + 1];
        float4 x2 = A4[f + 4];
        float4 x3 = A4[f + 5];
        v16bf a;
        a[0]  = f2bf(x0.x); a[1]  = f2bf(x0.y); a[2]  = f2bf(x0.z); a[3]  = f2bf(x0.w);
        a[4]  = f2bf(x1.x); a[5]  = f2bf(x1.y); a[6]  = f2bf(x1.z); a[7]  = f2bf(x1.w);
        a[8]  = f2bf(x2.x); a[9]  = f2bf(x2.y); a[10] = f2bf(x2.z); a[11] = f2bf(x2.w);
        a[12] = f2bf(x3.x); a[13] = f2bf(x3.y); a[14] = f2bf(x3.z); a[15] = f2bf(x3.w);
        v16bf b = Bp[kt4 * 256];
        c = __builtin_amdgcn_wmma_f32_16x16x32_bf16(
                false, a, false, b, (short)0, c, false, false);
    }
}

template <int NT>
__global__ void __launch_bounds__(256)
gemm_wmma(const float* __restrict__ A0, const float* __restrict__ W0, int K0,
          const float* __restrict__ A1, const float* __restrict__ W1, int K1,
          const float* __restrict__ A2, const float* __restrict__ W2, int K2,
          const float* __restrict__ bias,
          float* __restrict__ out, int doRelu) {
    __shared__ __bf16 Ws[HD * HD];            // 32 KB, swizzled B-fragment order

    const int lane = threadIdx.x & 31;
    const int wave = threadIdx.x >> 5;
    const int row  = blockIdx.x * 16 + (lane & 15);   // grid sized so row < N

    v8f c = {};

    stage_w(W0, K0, Ws);
    __syncthreads();
    gemm_term(A0, Ws, c, lane, wave, row);

    if constexpr (NT >= 2) {
        __syncthreads();
        stage_w(W1, K1, Ws);
        __syncthreads();
        gemm_term(A1, Ws, c, lane, wave, row);
    }
    if constexpr (NT >= 3) {
        __syncthreads();
        stage_w(W2, K2, Ws);
        __syncthreads();
        gemm_term(A2, Ws, c, lane, wave, row);
    }

    // D layout: vgpr r -> m = r + 8*(lane>=16); n = lane&15
    const int n  = wave * 16 + (lane & 15);
    const int m0 = blockIdx.x * 16 + ((lane >> 4) << 3);
    const float bv = bias[n];
#pragma unroll
    for (int r = 0; r < 8; ++r) {
        float v = c[r] + bv;
        if (doRelu) v = fmaxf(v, 0.0f);
        out[(long long)(m0 + r) * HD + n] = v;
    }
}

// ---------------------------------------------------------------------------
// segment_max pool via monotone uint keys + atomicMax
// ---------------------------------------------------------------------------
__global__ void pool_max(const float* __restrict__ xF,
                         const int* __restrict__ batch,
                         unsigned* __restrict__ pooled_u) {
    int f = blockIdx.x;
    int c = threadIdx.x;     // 128
    int g = batch[f];        // scalar
    float v = xF[(long long)f * HD + c];
    unsigned u; __builtin_memcpy(&u, &v, 4);
    unsigned key = (u & 0x80000000u) ? ~u : (u | 0x80000000u);
    atomicMax(&pooled_u[g * HD + c], key);
}

// ---------------------------------------------------------------------------
// Classifier: [64x128] -> relu@Wc1 [64x64] -> relu@Wc2 [64x128] -> @Wc3 [64x10]
// ---------------------------------------------------------------------------
__global__ void __launch_bounds__(256)
classifier(const unsigned* __restrict__ pooled_u,
           const float* __restrict__ Wc1, const float* __restrict__ bc1,
           const float* __restrict__ Wc2, const float* __restrict__ bc2,
           const float* __restrict__ Wc3, const float* __restrict__ bc3,
           float* __restrict__ out) {
    __shared__ float bufA[GB * HD];           // pooled, then h2
    __shared__ float bufB[GB * (HD / 2)];     // h1
    int tid = threadIdx.x;

    for (int i = tid; i < GB * HD; i += 256) {
        unsigned k = pooled_u[i];
        unsigned u = (k & 0x80000000u) ? (k & 0x7FFFFFFFu) : ~k;
        float f; __builtin_memcpy(&f, &u, 4);
        bufA[i] = f;
    }
    __syncthreads();

    for (int i = tid; i < GB * 64; i += 256) {        // h1 = relu(p @ Wc1 + b1)
        int g = i >> 6, c = i & 63;
        float acc = bc1[c];
        for (int k = 0; k < HD; ++k) acc += bufA[g * HD + k] * Wc1[k * 64 + c];
        bufB[i] = fmaxf(acc, 0.0f);
    }
    __syncthreads();

    for (int i = tid; i < GB * HD; i += 256) {        // h2 = relu(h1 @ Wc2 + b2)
        int g = i >> 7, c = i & 127;
        float acc = bc2[c];
        for (int k = 0; k < 64; ++k) acc += bufB[g * 64 + k] * Wc2[k * HD + c];
        bufA[i] = fmaxf(acc, 0.0f);   // bufA no longer read as "pooled"
    }
    __syncthreads();

    for (int i = tid; i < GB * NC; i += 256) {        // out = h2 @ Wc3 + b3
        int g = i / NC, c = i % NC;
        float acc = bc3[c];
        for (int k = 0; k < HD; ++k) acc += bufA[g * HD + k] * Wc3[k * NC + c];
        out[i] = acc;
    }
}

// ---------------------------------------------------------------------------
// Host orchestration
// ---------------------------------------------------------------------------
extern "C" void kernel_launch(void* const* d_in, const int* in_sizes, int n_in,
                              void* d_out, int out_size, void* d_ws, size_t ws_size,
                              hipStream_t stream) {
    (void)in_sizes; (void)n_in; (void)out_size; (void)ws_size;

    const int*   host_ids   = (const int*)  d_in[0];
    const float* flow_x     = (const float*)d_in[1];   // [N_FLOW x 97]
    const int*   h2f_src    = (const int*)  d_in[2];
    const int*   h2f_dst    = (const int*)  d_in[3];
    const int*   f2h_src    = (const int*)  d_in[4];
    const int*   f2h_dst    = (const int*)  d_in[5];
    const int*   rel_src    = (const int*)  d_in[6];
    const int*   rel_dst    = (const int*)  d_in[7];
    const int*   flow_batch = (const int*)  d_in[8];
    const float* host_embed = (const float*)d_in[9];   // [4000 x 128]
    const float* Wr0_h2f    = (const float*)d_in[10];  // [128 x 128]
    const float* br0_h2f    = (const float*)d_in[11];
    const float* Wo0_h2f    = (const float*)d_in[12];  // [97 x 128]
    const float* Wr0_f2h    = (const float*)d_in[13];  // [97 x 128]
    const float* br0_f2h    = (const float*)d_in[14];
    const float* Wo0_f2h    = (const float*)d_in[15];  // [128 x 128]
    const float* Wr0_rel    = (const float*)d_in[16];  // [97 x 128]
    const float* br0_rel    = (const float*)d_in[17];
    const float* Wo0_rel    = (const float*)d_in[18];  // [97 x 128]
    const float* Wr         = (const float*)d_in[19];  // [2,3,128,128]
    const float* br         = (const float*)d_in[20];  // [2,3,128]
    const float* Wo         = (const float*)d_in[21];  // [2,3,128,128]
    const float* Wc1        = (const float*)d_in[22];
    const float* bc1        = (const float*)d_in[23];
    const float* Wc2        = (const float*)d_in[24];
    const float* bc2        = (const float*)d_in[25];
    const float* Wc3        = (const float*)d_in[26];
    const float* bc3        = (const float*)d_in[27];
    float* out = (float*)d_out;

    // workspace carve-out
    char* ws = (char*)d_ws;
    auto carve = [&](size_t bytes) -> void* {
        void* p = (void*)ws;
        ws += (bytes + 255) & ~(size_t)255;
        return p;
    };
    float*    xF_a     = (float*)   carve((size_t)N_FLOW * HD * 4);
    float*    xF_b     = (float*)   carve((size_t)N_FLOW * HD * 4);  // layer0: padded flow_x
    float*    aggF1    = (float*)   carve((size_t)N_FLOW * HD * 4);
    float*    aggF2    = (float*)   carve((size_t)N_FLOW * HD * 4);
    float*    xH_a     = (float*)   carve((size_t)N_HOST * HD * 4);
    float*    xH_b     = (float*)   carve((size_t)N_HOST * HD * 4);
    float*    aggH     = (float*)   carve((size_t)N_HOST * HD * 4);
    float*    Wcomb    = (float*)   carve((size_t)HD * HD * 4);
    float*    bcomb    = (float*)   carve((size_t)HD * 4);
    unsigned* pooled_u = (unsigned*)carve((size_t)GB * HD * 4);

    auto zero_f = [&](float* p, long long n) {
        fill_zero_f<<<(unsigned)((n + 255) / 256), 256, 0, stream>>>(p, n);
    };

    const int gemmBlocksF = N_FLOW / 16;   // 12500 (exact)
    const int gemmBlocksH = N_HOST / 16;   // 1250  (exact)
    #define WR(l, t) (Wr + ((size_t)(l) * 3 + (t)) * HD * HD)
    #define WO(l, t) (Wo + ((size_t)(l) * 3 + (t)) * HD * HD)
    #define BR(l, t) (br + ((size_t)(l) * 3 + (t)) * HD)

    // ---- embedding lookup + padded flow features -------------------------
    gather_embed<<<N_HOST, 128, 0, stream>>>(host_embed, host_ids, xH_a);
    pad_flow<<<N_FLOW, 128, 0, stream>>>(flow_x, xF_b);   // xF_b = flow_x zero-padded

    // ======================= Layer 0 (hetero dims) ========================
    zero_f(aggF1, (long long)N_FLOW * HD);
    zero_f(aggF2, (long long)N_FLOW * HD);
    zero_f(aggH,  (long long)N_HOST * HD);
    scatter_add<<<E_HF,  128, 0, stream>>>(xH_a,   HD, h2f_src, h2f_dst, HD, aggF1);
    scatter_add<<<E_REL, 128, 0, stream>>>(flow_x, FI, rel_src, rel_dst, FI, aggF2);
    scatter_add<<<E_HF,  128, 0, stream>>>(flow_x, FI, f2h_src, f2h_dst, FI, aggH);
    add2_f<<<(FI * HD + 255) / 256, 256, 0, stream>>>(Wo0_h2f, Wo0_rel, Wcomb, FI * HD);
    add2_f<<<1, 128, 0, stream>>>(br0_h2f, br0_rel, bcomb, HD);
    // x_flow' = relu(aggF1@Wr0_h2f + aggF2@Wr0_rel + pad(flow_x)@(Wo0_h2f+Wo0_rel) + b)
    gemm_wmma<3><<<gemmBlocksF, 256, 0, stream>>>(
        aggF1, Wr0_h2f, HD, aggF2, Wr0_rel, FI, xF_b, Wcomb, FI,
        bcomb, xF_a, 1);
    // x_host' = relu(aggH@Wr0_f2h + x_host@Wo0_f2h + br0_f2h)
    gemm_wmma<2><<<gemmBlocksH, 256, 0, stream>>>(
        aggH, Wr0_f2h, FI, xH_a, Wo0_f2h, HD, nullptr, nullptr, 0,
        br0_f2h, xH_b, 1);

    // ======================= Layer 1 (relu) ===============================
    zero_f(aggF1, (long long)N_FLOW * HD);
    zero_f(aggF2, (long long)N_FLOW * HD);
    zero_f(aggH,  (long long)N_HOST * HD);
    scatter_add<<<E_HF,  128, 0, stream>>>(xH_b, HD, h2f_src, h2f_dst, HD, aggF1);
    scatter_add<<<E_REL, 128, 0, stream>>>(xF_a, HD, rel_src, rel_dst, HD, aggF2);
    scatter_add<<<E_HF,  128, 0, stream>>>(xF_a, HD, f2h_src, f2h_dst, HD, aggH);
    add2_f<<<(HD * HD + 255) / 256, 256, 0, stream>>>(WO(0, 0), WO(0, 2), Wcomb, HD * HD);
    add2_f<<<1, 128, 0, stream>>>(BR(0, 0), BR(0, 2), bcomb, HD);
    gemm_wmma<3><<<gemmBlocksF, 256, 0, stream>>>(
        aggF1, WR(0, 0), HD, aggF2, WR(0, 2), HD, xF_a, Wcomb, HD,
        bcomb, xF_b, 1);
    gemm_wmma<2><<<gemmBlocksH, 256, 0, stream>>>(
        aggH, WR(0, 1), HD, xH_b, WO(0, 1), HD, nullptr, nullptr, 0,
        BR(0, 1), xH_a, 1);

    // ============ Layer 2 (no relu; host update is dead code) =============
    zero_f(aggF1, (long long)N_FLOW * HD);
    zero_f(aggF2, (long long)N_FLOW * HD);
    scatter_add<<<E_HF,  128, 0, stream>>>(xH_a, HD, h2f_src, h2f_dst, HD, aggF1);
    scatter_add<<<E_REL, 128, 0, stream>>>(xF_b, HD, rel_src, rel_dst, HD, aggF2);
    add2_f<<<(HD * HD + 255) / 256, 256, 0, stream>>>(WO(1, 0), WO(1, 2), Wcomb, HD * HD);
    add2_f<<<1, 128, 0, stream>>>(BR(1, 0), BR(1, 2), bcomb, HD);
    gemm_wmma<3><<<gemmBlocksF, 256, 0, stream>>>(
        aggF1, WR(1, 0), HD, aggF2, WR(1, 2), HD, xF_b, Wcomb, HD,
        bcomb, xF_a, 0);

    // ===================== global max pool + classifier ===================
    fill_zero_u<<<(GB * HD + 255) / 256, 256, 0, stream>>>(pooled_u, GB * HD);
    pool_max<<<N_FLOW, 128, 0, stream>>>(xF_a, flow_batch, pooled_u);
    classifier<<<1, 256, 0, stream>>>(pooled_u, Wc1, bc1, Wc2, bc2, Wc3, bc3, out);
}
